// BlockDrop_82076825026985
// MI455X (gfx1250) — compile-verified, matrix operand order checked by reading, not run on hardware
//
#include <hip/hip_runtime.h>
#include <cstdint>
#include <cstddef>

// ---------------------------------------------------------------------------
// Types for CDNA5 WMMA (gfx1250, wave32)
// ---------------------------------------------------------------------------
typedef __bf16 bf16_t;
typedef __attribute__((ext_vector_type(16))) __bf16 v16bf;
typedef __attribute__((ext_vector_type(8)))  __bf16 v8bf;
typedef __attribute__((ext_vector_type(8)))  float  v8f;

union V16U { v16bf v; v8bf h[2]; };

__device__ __forceinline__ uint32_t bfbits(float f) {       // RNE fp32->bf16 bits
  uint32_t u = __float_as_uint(f);
  return (u + 0x7FFFu + ((u >> 16) & 1u)) >> 16;
}
__device__ __forceinline__ bf16_t f2bf(float f) {
  uint16_t s = (uint16_t)bfbits(f);
  return __builtin_bit_cast(bf16_t, s);
}

// ---------------------------------------------------------------------------
// CDNA5 async global->LDS staging (GLOBAL_LOAD_ASYNC_TO_LDS_B128, ASYNCcnt).
// Generic LDS pointer low 32 bits == LDS byte address (ISA 10.2); generic
// global pointer == global address. IOFFSET applies to BOTH addresses.
// ---------------------------------------------------------------------------
template <int OFF>
__device__ __forceinline__ void async_copy16(const void* g, void* l) {
  asm volatile("global_load_async_to_lds_b128 %0, %1, off offset:%2"
               :: "v"((uint32_t)(uintptr_t)l),
                  "v"((uint64_t)(uintptr_t)g),
                  "i"(OFF)
               : "memory");
}
__device__ __forceinline__ void wait_async0() {
  asm volatile("s_wait_asynccnt 0x0" ::: "memory");
}

// Load a 16x32 bf16 operand fragment from a row-major tile.
// A-operand: lane = M row.  B-operand (symmetric layout): lane = N column,
// tile must hold that column's K-run contiguously.
// ISA: lanes 0-15 take K {0..7,16..23}; lanes 16-31 take K {8..15,24..31}.
__device__ __forceinline__ v16bf load_frag(const bf16_t* __restrict__ base,
                                           int row_stride, int lane) {
  int m    = lane & 15;
  int half = lane >> 4;
  const bf16_t* r = base + (size_t)m * row_stride;
  V16U u;
  u.h[0] = *(const v8bf*)(r + half * 8);
  u.h[1] = *(const v8bf*)(r + 16 + half * 8);
  return u.v;
}

__device__ __forceinline__ v8f wmma_bf16(v16bf a, v16bf b, v8f c) {
  return __builtin_amdgcn_wmma_f32_16x16x32_bf16(
      false, a, false, b, (short)0, c, false, false);
}

// ---------------------------------------------------------------------------
// Problem constants
// ---------------------------------------------------------------------------
#define DIM   1024
#define HEADS 16
#define HDIM  64
#define HID   4096
#define ROWS  32768
#define NWIN  128
#define NTOK  256
#define CHUNK 4096

__device__ __forceinline__ int permute_row(int row) {     // natural -> window
  int b = row >> 12, rem = row & 4095;
  int y = rem >> 6, xc = rem & 63;
  int win = (b << 4) + ((y >> 4) << 2) + (xc >> 4);
  return (win << 8) + ((y & 15) << 4) + (xc & 15);
}
__device__ __forceinline__ int unpermute_row(int wrow) {  // window -> natural
  int win = wrow >> 8, pos = wrow & 255;
  int b = win >> 4, wy = (win >> 2) & 3, wx = win & 3;
  int y = wy * 16 + (pos >> 4);
  int xc = wx * 16 + (pos & 15);
  return (b << 12) + (y << 6) + xc;
}

// ---------------------------------------------------------------------------
// Weight convert+transpose: W[K,N] f32 -> Wt[N,K] bf16 (done ONCE, so the
// GEMM hot loop stages B tiles with pure async b128 copies, no VALU).
// 32x32 tile through LDS; one block = one tile.
// ---------------------------------------------------------------------------
__global__ __launch_bounds__(256) void wcvt_kernel(
    const float* __restrict__ W, bf16_t* __restrict__ Wt, int K, int N) {
  __shared__ float T[32][33];
  int k0 = blockIdx.x * 32, n0 = blockIdx.y * 32;
  int t = threadIdx.x;
  int r = t >> 3, c4 = (t & 7) * 4;
  float4 f = *(const float4*)(W + (size_t)(k0 + r) * N + n0 + c4);
  T[r][c4 + 0] = f.x; T[r][c4 + 1] = f.y;
  T[r][c4 + 2] = f.z; T[r][c4 + 3] = f.w;
  __syncthreads();
  uint32_t lo = bfbits(T[c4 + 0][r]) | (bfbits(T[c4 + 1][r]) << 16);
  uint32_t hi = bfbits(T[c4 + 2][r]) | (bfbits(T[c4 + 3][r]) << 16);
  uint2* d = (uint2*)(Wt + (size_t)(n0 + r) * K + k0 + c4);
  *d = make_uint2(lo, hi);
}

// ---------------------------------------------------------------------------
// LayerNorm (+ optional window permutation), fp32 -> bf16
// ---------------------------------------------------------------------------
__global__ __launch_bounds__(256) void ln_kernel(
    const float* __restrict__ x, const float* __restrict__ gamma,
    const float* __restrict__ beta, bf16_t* __restrict__ out, int permute) {
  int row = blockIdx.x;
  int t = threadIdx.x;
  const float* xr = x + (size_t)row * DIM;

  float vals[4], s = 0.f, ss = 0.f;
#pragma unroll
  for (int i = 0; i < 4; ++i) {
    float v = xr[t + 256 * i];
    vals[i] = v; s += v; ss += v * v;
  }
#pragma unroll
  for (int m = 16; m >= 1; m >>= 1) {
    s  += __shfl_xor(s,  m, 32);
    ss += __shfl_xor(ss, m, 32);
  }
  __shared__ float as_[8], bs_[8];
  if ((t & 31) == 0) { as_[t >> 5] = s; bs_[t >> 5] = ss; }
  __syncthreads();
  float S = 0.f, SS = 0.f;
#pragma unroll
  for (int i = 0; i < 8; ++i) { S += as_[i]; SS += bs_[i]; }
  float mean = S * (1.f / DIM);
  float var  = SS * (1.f / DIM) - mean * mean;
  float rstd = rsqrtf(var + 1e-5f);

  int orow = permute ? permute_row(row) : row;
  bf16_t* op = out + (size_t)orow * DIM;
#pragma unroll
  for (int i = 0; i < 4; ++i) {
    int c = t + 256 * i;
    op[c] = f2bf((vals[i] - mean) * rstd * gamma[c] + beta[c]);
  }
}

// ---------------------------------------------------------------------------
// WMMA GEMM: out[M,N] = A[M,K](bf16) * Wt[N,K](bf16, pre-transposed) + bias
// Block tile 128x128, K-step 32, double-buffered LDS; BOTH operands staged
// with GLOBAL_LOAD_ASYNC_TO_LDS_B128. 8 waves (4x2); wave = 32x64 = 2x4 WMMA.
// mode 0: bf16 (QKV) | 1: un-permute+residual f32 (proj) |
// mode 2: GELU bf16 (fc1) | 3: residual f32 (fc2 -> d_out)
// ---------------------------------------------------------------------------
#define BM 128
#define BN 128
#define BK 32

__global__ __launch_bounds__(256) void gemm_kernel(
    const bf16_t* __restrict__ A, const bf16_t* __restrict__ Wt,
    const float* __restrict__ bias, void* __restrict__ outp,
    const float* __restrict__ residual, int M, int N, int K, int mode) {
  __shared__ bf16_t As[2][BM][BK];
  __shared__ bf16_t Bs[2][BN][BK];   // Bs[n][k]

  int tid = threadIdx.x;
  int lane = tid & 31, wave = tid >> 5;
  int wm = wave >> 1, wn = wave & 1;
  int m0 = blockIdx.y * BM, n0 = blockIdx.x * BN;

  v8f acc[2][4];
#pragma unroll
  for (int i = 0; i < 2; ++i)
#pragma unroll
    for (int j = 0; j < 4; ++j)
      acc[i][j] = v8f{0.f, 0.f, 0.f, 0.f, 0.f, 0.f, 0.f, 0.f};

  int tr = tid >> 1, tc = (tid & 1) * 16;   // 32B per thread for each tile
  const bf16_t* arow = A  + (size_t)(m0 + tr) * K + tc;
  const bf16_t* brow = Wt + (size_t)(n0 + tr) * K + tc;

  const int nsteps = K / BK;

  // prologue: stage k-step 0 into buffer 0 (4 async b128 per thread)
  async_copy16<0>(arow, &As[0][tr][tc]);
  async_copy16<16>(arow, &As[0][tr][tc]);
  async_copy16<0>(brow, &Bs[0][tr][tc]);
  async_copy16<16>(brow, &Bs[0][tr][tc]);

  for (int s = 0; s < nsteps; ++s) {
    int cur = s & 1, nxt = cur ^ 1;
    wait_async0();          // own async stores into buf[cur] have landed
    __syncthreads();        // all waves done writing buf[cur] / reading buf[nxt]

    if (s + 1 < nsteps) {
      int k0 = (s + 1) * BK;
      async_copy16<0>(arow + k0, &As[nxt][tr][tc]);
      async_copy16<16>(arow + k0, &As[nxt][tr][tc]);
      async_copy16<0>(brow + k0, &Bs[nxt][tr][tc]);
      async_copy16<16>(brow + k0, &Bs[nxt][tr][tc]);
      if (s + 2 < nsteps) {
        __builtin_prefetch(arow + k0 + BK, 0, 3);
        __builtin_prefetch(brow + k0 + BK, 0, 3);
      }
    }

    // compute from buf[cur]
    v16bf a0 = load_frag(&As[cur][wm * 32 + 0][0],  BK, lane);
    v16bf a1 = load_frag(&As[cur][wm * 32 + 16][0], BK, lane);
#pragma unroll
    for (int j = 0; j < 4; ++j) {
      v16bf b = load_frag(&Bs[cur][wn * 64 + j * 16][0], BK, lane);
      acc[0][j] = wmma_bf16(a0, b, acc[0][j]);
      acc[1][j] = wmma_bf16(a1, b, acc[1][j]);
    }
  }

  // epilogue
  int half = lane >> 4, ln = lane & 15;
#pragma unroll
  for (int i = 0; i < 2; ++i)
#pragma unroll
    for (int j = 0; j < 4; ++j)
#pragma unroll
      for (int r = 0; r < 8; ++r) {
        int row = m0 + wm * 32 + i * 16 + half * 8 + r;
        int col = n0 + wn * 64 + j * 16 + ln;
        float v = acc[i][j][r] + bias[col];
        if (mode == 0) {
          ((bf16_t*)outp)[(size_t)row * N + col] = f2bf(v);
        } else if (mode == 1) {
          int nat = unpermute_row(row);
          size_t idx = (size_t)nat * N + col;
          ((float*)outp)[idx] = residual[idx] + v;
        } else if (mode == 2) {
          float g = 0.5f * v * (1.f + erff(v * 0.70710678118654752f));
          ((bf16_t*)outp)[(size_t)row * N + col] = f2bf(g);
        } else {
          size_t idx = (size_t)row * N + col;
          ((float*)outp)[idx] = residual[idx] + v;
        }
      }
}

// ---------------------------------------------------------------------------
// Fused window attention. One block per (window, head): 512 threads = 16 waves.
// Wave w owns queries [16w,16w+16); 16x256 score strip lives in 16 WMMA f32
// accumulators; softmax via in-lane + half-wave __shfl_xor reductions.
// ---------------------------------------------------------------------------
__global__ __launch_bounds__(512, 1) void attn_kernel(
    const bf16_t* __restrict__ q, const bf16_t* __restrict__ k,
    const bf16_t* __restrict__ v, bf16_t* __restrict__ o) {
  extern __shared__ char smem[];
  bf16_t* Ks = (bf16_t*)smem;            // [256][64] keys x dim
  bf16_t* Vt = Ks + NTOK * HDIM;         // [64][256] dim x key (V^T)
  bf16_t* Ps = Vt + HDIM * NTOK;         // 16 waves x [16][32]

  int tid = threadIdx.x, lane = tid & 31, wave = tid >> 5;
  int win = blockIdx.x >> 4, head = blockIdx.x & 15;
  size_t base = (size_t)win * NTOK * DIM + (size_t)head * HDIM;

  // stage K (async, row-major) and V (transposed)
  {
    int r = tid >> 1, c = (tid & 1) * 32;
    const bf16_t* ksrc = k + base + (size_t)r * DIM + c;
    bf16_t* kdst = Ks + r * HDIM + c;
    async_copy16<0>(ksrc, kdst);
    async_copy16<16>(ksrc, kdst);
    async_copy16<32>(ksrc, kdst);
    async_copy16<48>(ksrc, kdst);
    const bf16_t* vsrc = v + base + (size_t)r * DIM + c;
#pragma unroll
    for (int i = 0; i < 32; ++i) Vt[(size_t)(c + i) * NTOK + r] = vsrc[i];
  }

  // Q fragments straight from global (A layout)
  int q0 = wave * 16;
  const bf16_t* qrow = q + base + (size_t)q0 * DIM;
  v16bf aq0 = load_frag(qrow, DIM, lane);
  v16bf aq1 = load_frag(qrow + 32, DIM, lane);
  wait_async0();
  __syncthreads();

  // S = Q K^T
  v8f s[16];
#pragma unroll
  for (int j = 0; j < 16; ++j)
    s[j] = v8f{0.f, 0.f, 0.f, 0.f, 0.f, 0.f, 0.f, 0.f};
#pragma unroll
  for (int j = 0; j < 16; ++j) {
    v16bf b0 = load_frag(Ks + (size_t)(j * 16) * HDIM, HDIM, lane);
    v16bf b1 = load_frag(Ks + (size_t)(j * 16) * HDIM + 32, HDIM, lane);
    s[j] = wmma_bf16(aq0, b0, s[j]);
    s[j] = wmma_bf16(aq1, b1, s[j]);
  }

  // softmax (scale = 0.125)
  const float SC = 0.125f;
  float rmax[8], rinv[8];
#pragma unroll
  for (int r = 0; r < 8; ++r) {
    float m = -1e30f;
#pragma unroll
    for (int j = 0; j < 16; ++j) m = fmaxf(m, s[j][r]);
#pragma unroll
    for (int d = 8; d >= 1; d >>= 1) m = fmaxf(m, __shfl_xor(m, d, 32));
    rmax[r] = m * SC;
  }
#pragma unroll
  for (int j = 0; j < 16; ++j)
#pragma unroll
    for (int r = 0; r < 8; ++r)
      s[j][r] = __expf(s[j][r] * SC - rmax[r]);
#pragma unroll
  for (int r = 0; r < 8; ++r) {
    float t = 0.f;
#pragma unroll
    for (int j = 0; j < 16; ++j) t += s[j][r];
#pragma unroll
    for (int d = 8; d >= 1; d >>= 1) t += __shfl_xor(t, d, 32);
    rinv[r] = 1.f / t;
  }
#pragma unroll
  for (int j = 0; j < 16; ++j)
#pragma unroll
    for (int r = 0; r < 8; ++r)
      s[j][r] *= rinv[r];

  // O = P V, re-layout P via per-wave LDS scratch
  bf16_t* myP = Ps + (size_t)wave * 16 * 32;
  int half = lane >> 4, ln = lane & 15;
  v8f oacc[4];
#pragma unroll
  for (int jd = 0; jd < 4; ++jd)
    oacc[jd] = v8f{0.f, 0.f, 0.f, 0.f, 0.f, 0.f, 0.f, 0.f};

  for (int kb = 0; kb < 8; ++kb) {
#pragma unroll
    for (int jj = 0; jj < 2; ++jj) {
      int j = kb * 2 + jj;
#pragma unroll
      for (int r = 0; r < 8; ++r)
        myP[(half * 8 + r) * 32 + jj * 16 + ln] = f2bf(s[j][r]);
    }
    v16bf ap = load_frag(myP, 32, lane);   // per-wave private; LDS in-order
#pragma unroll
    for (int jd = 0; jd < 4; ++jd) {
      v16bf bv = load_frag(Vt + (size_t)(jd * 16) * NTOK + kb * 32, NTOK, lane);
      oacc[jd] = wmma_bf16(ap, bv, oacc[jd]);
    }
  }

  // store O
  bf16_t* orow = o + base + (size_t)q0 * DIM;
#pragma unroll
  for (int jd = 0; jd < 4; ++jd)
#pragma unroll
    for (int r = 0; r < 8; ++r)
      orow[(size_t)(half * 8 + r) * DIM + jd * 16 + ln] = f2bf(oacc[jd][r]);
}

// ---------------------------------------------------------------------------
// Launch. Workspace plan (280 MiB total):
//   [0,64Mi)      h -> o -> MLP hidden chunk
//   [64,192Mi)    q,k -> x1 (f32)
//   [192,256Mi)   v -> h2
//   [256,280Mi)   bf16 transposed weights (Wq,Wk,Wv,Wo: 2Mi each; W1,W2: 8Mi)
// ---------------------------------------------------------------------------
extern "C" void kernel_launch(void* const* d_in, const int* in_sizes, int n_in,
                              void* d_out, int out_size, void* d_ws, size_t ws_size,
                              hipStream_t stream) {
  const float* x    = (const float*)d_in[0];
  const float* g1   = (const float*)d_in[1];
  const float* be1  = (const float*)d_in[2];
  const float* Wq   = (const float*)d_in[3];
  const float* bq   = (const float*)d_in[4];
  const float* Wk   = (const float*)d_in[5];
  const float* bk   = (const float*)d_in[6];
  const float* Wv   = (const float*)d_in[7];
  const float* bv   = (const float*)d_in[8];
  const float* Wo   = (const float*)d_in[9];
  const float* bo   = (const float*)d_in[10];
  const float* g2   = (const float*)d_in[11];
  const float* be2  = (const float*)d_in[12];
  const float* W1   = (const float*)d_in[13];
  const float* b1m  = (const float*)d_in[14];
  const float* W2   = (const float*)d_in[15];
  const float* b2m  = (const float*)d_in[16];
  float* out = (float*)d_out;

  char* w = (char*)d_ws;
  const size_t MiB = 1u << 20;
  bf16_t* hbuf = (bf16_t*)(w);
  bf16_t* qb   = (bf16_t*)(w + 64 * MiB);
  bf16_t* kb   = (bf16_t*)(w + 128 * MiB);
  bf16_t* vb   = (bf16_t*)(w + 192 * MiB);
  bf16_t* ob   = hbuf;
  float*  x1   = (float*)(w + 64 * MiB);
  bf16_t* h2   = (bf16_t*)(w + 192 * MiB);
  bf16_t* ub   = (bf16_t*)(w);
  bf16_t* wtq  = (bf16_t*)(w + 256 * MiB);
  bf16_t* wtk  = (bf16_t*)(w + 258 * MiB);
  bf16_t* wtv  = (bf16_t*)(w + 260 * MiB);
  bf16_t* wto  = (bf16_t*)(w + 262 * MiB);
  bf16_t* wt1  = (bf16_t*)(w + 264 * MiB);
  bf16_t* wt2  = (bf16_t*)(w + 272 * MiB);

  // 0. one-time weight convert+transpose (f32 [K,N] -> bf16 [N,K])
  dim3 gw(DIM / 32, DIM / 32);
  wcvt_kernel<<<gw, 256, 0, stream>>>(Wq, wtq, DIM, DIM);
  wcvt_kernel<<<gw, 256, 0, stream>>>(Wk, wtk, DIM, DIM);
  wcvt_kernel<<<gw, 256, 0, stream>>>(Wv, wtv, DIM, DIM);
  wcvt_kernel<<<gw, 256, 0, stream>>>(Wo, wto, DIM, DIM);
  wcvt_kernel<<<dim3(DIM / 32, HID / 32), 256, 0, stream>>>(W1, wt1, DIM, HID);
  wcvt_kernel<<<dim3(HID / 32, DIM / 32), 256, 0, stream>>>(W2, wt2, HID, DIM);

  // 1. LN1 + window partition
  ln_kernel<<<ROWS, 256, 0, stream>>>(x, g1, be1, hbuf, 1);

  // 2. QKV projections
  dim3 gq(DIM / BN, ROWS / BM);
  gemm_kernel<<<gq, 256, 0, stream>>>(hbuf, wtq, bq, qb, nullptr, ROWS, DIM, DIM, 0);
  gemm_kernel<<<gq, 256, 0, stream>>>(hbuf, wtk, bk, kb, nullptr, ROWS, DIM, DIM, 0);
  gemm_kernel<<<gq, 256, 0, stream>>>(hbuf, wtv, bv, vb, nullptr, ROWS, DIM, DIM, 0);

  // 3. fused window attention
  attn_kernel<<<NWIN * HEADS, 512, 81920, stream>>>(qb, kb, vb, ob);

  // 4. output projection + un-permute + residual
  gemm_kernel<<<gq, 256, 0, stream>>>(ob, wto, bo, x1, x, ROWS, DIM, DIM, 1);

  // 5. LN2
  ln_kernel<<<ROWS, 256, 0, stream>>>(x1, g2, be2, h2, 0);

  // 6. MLP in 8 row-chunks
  for (int c = 0; c < 8; ++c) {
    size_t off = (size_t)c * CHUNK;
    dim3 g1d(HID / BN, CHUNK / BM);
    gemm_kernel<<<g1d, 256, 0, stream>>>(h2 + off * DIM, wt1, b1m, ub, nullptr,
                                         CHUNK, HID, DIM, 2);
    dim3 g2d(DIM / BN, CHUNK / BM);
    gemm_kernel<<<g2d, 256, 0, stream>>>(ub, wt2, b2m, out + off * DIM,
                                         x1 + off * DIM, CHUNK, DIM, HID, 3);
  }
}
